// OptimizedMoELayer_4148938408538
// MI455X (gfx1250) — matrix operand order repaired and checked by reading, not compiled, not run on hardware
//
#include <hip/hip_runtime.h>
#include <hip/hip_bf16.h>

typedef __attribute__((ext_vector_type(16))) __bf16 v16bf;
typedef __attribute__((ext_vector_type(8)))  __bf16 v8bf;
typedef __attribute__((ext_vector_type(8)))  float  v8f;

namespace {
constexpr int Bb = 2, Ss = 1024, Hh = 768, Ii = 1536, Ee = 8, Kk = 2;
constexpr int T    = Bb * Ss;                       // 2048 tokens
constexpr int CAP  = 768;                           // max(8, int((T*K//E)*1.5))
constexpr int POOL = 4608;                          // >= sum_e ceil(kept_e/64)*64
constexpr int TM = 64, TN = 128, TK = 32;
constexpr int MAXMT = (2 * CAP + TM - 1) / TM;      // 24 m-tiles max per expert
}

// ---------------------------------------------------------------------------
// 1) Routing: one wave per token. logits = x@w_gate, softmax, top-2, norm+clamp
// ---------------------------------------------------------------------------
__global__ __launch_bounds__(256) void routing_kernel(
    const float* __restrict__ x, const float* __restrict__ wg,
    float* __restrict__ gate_probs, int* __restrict__ topk_idx,
    float* __restrict__ w_clamped)
{
  const int t    = blockIdx.x * 8 + (threadIdx.x >> 5);
  const int lane = threadIdx.x & 31;

  float acc[Ee];
#pragma unroll
  for (int e = 0; e < Ee; ++e) acc[e] = 0.f;

  const float* xr = x + (size_t)t * Hh;
  for (int h = lane; h < Hh; h += 32) {
    float xv = xr[h];
    const float* wr = wg + (size_t)h * Ee;
#pragma unroll
    for (int e = 0; e < Ee; ++e) acc[e] += xv * wr[e];
  }
#pragma unroll
  for (int off = 16; off > 0; off >>= 1)
#pragma unroll
    for (int e = 0; e < Ee; ++e) acc[e] += __shfl_xor(acc[e], off, 32);

  if (lane == 0) {
    float mx = acc[0];
#pragma unroll
    for (int e = 1; e < Ee; ++e) mx = fmaxf(mx, acc[e]);
    float p[Ee], s = 0.f;
#pragma unroll
    for (int e = 0; e < Ee; ++e) { p[e] = __expf(acc[e] - mx); s += p[e]; }
#pragma unroll
    for (int e = 0; e < Ee; ++e) { p[e] /= s; gate_probs[(size_t)t * Ee + e] = p[e]; }
    int i0 = 0;
#pragma unroll
    for (int e = 1; e < Ee; ++e) if (p[e] > p[i0]) i0 = e;
    int i1 = (i0 == 0) ? 1 : 0;
#pragma unroll
    for (int e = 0; e < Ee; ++e) if (e != i0 && p[e] > p[i1]) i1 = e;
    float den = p[i0] + p[i1] + 1e-8f;
    float c0 = fminf(fmaxf(p[i0] / den, 1e-8f), 10.f);
    float c1 = fminf(fmaxf(p[i1] / den, 1e-8f), 10.f);
    topk_idx[t * Kk + 0] = i0;  topk_idx[t * Kk + 1] = i1;
    w_clamped[t * Kk + 0] = c0; w_clamped[t * Kk + 1] = c1;
  }
}

// ---------------------------------------------------------------------------
// 2) Dispatch: per-(k,expert) token-order rank (cumsum axis=0), capacity,
//    compact pooled row assignment (per-expert regions rounded to 64 rows).
// ---------------------------------------------------------------------------
__global__ __launch_bounds__(256) void dispatch_kernel(
    const int* __restrict__ topk_idx, int* __restrict__ slot_row,
    int* __restrict__ token_pool, int* __restrict__ row_base,
    int* __restrict__ kept, float* __restrict__ droppedp)
{
  __shared__ int scnt[256][16];
  __shared__ int stot[16];
  __shared__ int soff[16];
  __shared__ int sbase[Ee];
  const int tid = threadIdx.x;

  int cnt[16];
#pragma unroll
  for (int j = 0; j < 16; ++j) cnt[j] = 0;
  for (int i = 0; i < 8; ++i) {
    int t = tid * 8 + i;
#pragma unroll
    for (int k = 0; k < Kk; ++k) cnt[k * 8 + topk_idx[t * Kk + k]]++;
  }
#pragma unroll
  for (int j = 0; j < 16; ++j) scnt[tid][j] = cnt[j];
  __syncthreads();

  if (tid < 16) {
    int run = 0;
    for (int i = 0; i < 256; ++i) { int v = scnt[i][tid]; scnt[i][tid] = run; run += v; }
    stot[tid] = run;
  }
  __syncthreads();

  if (tid == 0) {
    int b = 0; float drop = 0.f;
    for (int e = 0; e < Ee; ++e) {
      int k0c = stot[e]     < CAP ? stot[e]     : CAP;
      int k1c = stot[8 + e] < CAP ? stot[8 + e] : CAP;
      soff[e] = 0; soff[8 + e] = k0c;
      kept[e] = k0c + k1c;
      row_base[e] = b; sbase[e] = b;
      b += ((k0c + k1c + 63) / 64) * 64;
      if (stot[e]     > CAP) drop += (float)(stot[e]     - CAP);
      if (stot[8 + e] > CAP) drop += (float)(stot[8 + e] - CAP);
    }
    *droppedp = drop;
  }
  __syncthreads();

  int rank[16];
#pragma unroll
  for (int j = 0; j < 16; ++j) rank[j] = scnt[tid][j];
  for (int i = 0; i < 8; ++i) {
    int t = tid * 8 + i;
#pragma unroll
    for (int k = 0; k < Kk; ++k) {
      int e = topk_idx[t * Kk + k];
      int r = rank[k * 8 + e]++;
      if (r < CAP) {
        int row = sbase[e] + soff[k * 8 + e] + r;
        slot_row[t * Kk + k] = row;
        token_pool[row] = t;
      } else {
        slot_row[t * Kk + k] = -1;
      }
    }
  }
}

// ---------------------------------------------------------------------------
// 3a) x fp32 -> bf16 (row layout preserved: [T][H])
// ---------------------------------------------------------------------------
__global__ __launch_bounds__(256) void cvt_x_kernel(
    const float* __restrict__ src, __bf16* __restrict__ dst)
{
  const size_t i = ((size_t)blockIdx.x * 256 + threadIdx.x) * 8;
  float4 a = *(const float4*)(src + i);
  float4 b = *(const float4*)(src + i + 4);
  v8bf o;
  o[0] = (__bf16)a.x; o[1] = (__bf16)a.y; o[2] = (__bf16)a.z; o[3] = (__bf16)a.w;
  o[4] = (__bf16)b.x; o[5] = (__bf16)b.y; o[6] = (__bf16)b.z; o[7] = (__bf16)b.w;
  *(v8bf*)(dst + i) = o;
}

// ---------------------------------------------------------------------------
// 3b) Weight transpose + convert: src fp32 [E][R][C] -> dst bf16 [E][C][R]
//     (makes the WMMA K dimension contiguous per output column)
// ---------------------------------------------------------------------------
__global__ __launch_bounds__(256) void transpose_w_kernel(
    const float* __restrict__ src, __bf16* __restrict__ dst, int R, int C)
{
  __shared__ __bf16 Ls[64][80];
  const int e  = blockIdx.z;
  const int r0 = blockIdx.y * 64, c0 = blockIdx.x * 64;
  const int tid = threadIdx.x;
#pragma unroll
  for (int it = 0; it < 4; ++it) {
    int idx = tid + it * 256;
    int r = idx >> 4, c4 = idx & 15;
    float4 v = *(const float4*)(src + ((size_t)e * R + r0 + r) * C + c0 + c4 * 4);
    Ls[c4 * 4 + 0][r] = (__bf16)v.x; Ls[c4 * 4 + 1][r] = (__bf16)v.y;
    Ls[c4 * 4 + 2][r] = (__bf16)v.z; Ls[c4 * 4 + 3][r] = (__bf16)v.w;
  }
  __syncthreads();
  {
    int ci = tid >> 2, rq = tid & 3;
    v8bf lo = *(const v8bf*)&Ls[ci][rq * 16];
    v8bf hi = *(const v8bf*)&Ls[ci][rq * 16 + 8];
    __bf16* dp = dst + ((size_t)e * C + c0 + ci) * R + r0 + rq * 16;
    *(v8bf*)dp = lo;
    *(v8bf*)(dp + 8) = hi;
  }
}

// ---------------------------------------------------------------------------
// Fragment helpers: both A and B WMMA operands are per-lane K-contiguous, so
// they are direct 16-byte global loads from K-major bf16 buffers.
//   A frag (lane row = l%16, hf = l/16): K[hf*8 .. +8) and K[16+hf*8 .. +8)
//   B frag (lane col = l%16, hf = l/16): K[hf*8 .. +8) and K[16+hf*8 .. +8)
// ---------------------------------------------------------------------------
__device__ __forceinline__ v16bf cat8(v8bf lo, v8bf hi) {
  return __builtin_shufflevector(lo, hi, 0, 1, 2, 3, 4, 5, 6, 7,
                                 8, 9, 10, 11, 12, 13, 14, 15);
}
__device__ __forceinline__ v16bf load_a_frag_g(const __bf16* rowp, int k0, int hf) {
  v8bf lo = *(const v8bf*)(rowp + k0 + hf * 8);
  v8bf hi = *(const v8bf*)(rowp + k0 + 16 + hf * 8);
  return cat8(lo, hi);
}
__device__ __forceinline__ v16bf load_b_frag_g(const __bf16* colp, int k0, int hf) {
  v8bf lo = *(const v8bf*)(colp + k0 + hf * 8);
  v8bf hi = *(const v8bf*)(colp + k0 + 16 + hf * 8);
  return cat8(lo, hi);
}

// ---------------------------------------------------------------------------
// 4) FFN1: G = Xg @ W1, U = Xg @ W3, hact = silu(G)*U (bf16). LDS-free loop.
// ---------------------------------------------------------------------------
__global__ __launch_bounds__(256) void ffn1_kernel(
    const __bf16* __restrict__ xbf, const __bf16* __restrict__ w1t,
    const __bf16* __restrict__ w3t, const int* __restrict__ token_pool,
    const int* __restrict__ row_base, const int* __restrict__ kept,
    __bf16* __restrict__ hact)
{
  const int e  = blockIdx.z;
  const int nk = kept[e];
  const int mt = blockIdx.y;
  if (mt * TM >= nk) return;
  const int base = row_base[e];
  const int n0 = blockIdx.x * TN;

  const int tid = threadIdx.x, lane = tid & 31, wv = tid >> 5;
  const int wm = wv >> 2, wn = wv & 3;
  const int hf = lane >> 4, l16 = lane & 15;

  // Hoisted per-lane A row gather and B column pointers
  const __bf16* arow[2];
  bool aval[2];
#pragma unroll
  for (int mi = 0; mi < 2; ++mi) {
    int m = mt * TM + wm * 32 + mi * 16 + l16;
    aval[mi] = (m < nk);
    int tok = aval[mi] ? token_pool[base + m] : 0;
    arow[mi] = xbf + (size_t)tok * Hh;
  }
  const __bf16 *bcol1[2], *bcol3[2];
#pragma unroll
  for (int ni = 0; ni < 2; ++ni) {
    int n = n0 + wn * 32 + ni * 16 + l16;
    bcol1[ni] = w1t + ((size_t)e * Ii + n) * Hh;
    bcol3[ni] = w3t + ((size_t)e * Ii + n) * Hh;
  }

  v8f accG[2][2], accU[2][2];
#pragma unroll
  for (int mi = 0; mi < 2; ++mi)
#pragma unroll
    for (int ni = 0; ni < 2; ++ni)
#pragma unroll
      for (int i = 0; i < 8; ++i) { accG[mi][ni][i] = 0.f; accU[mi][ni][i] = 0.f; }

  for (int k0 = 0; k0 < Hh; k0 += TK) {
    v16bf af[2], b1f[2], b3f[2];
#pragma unroll
    for (int mi = 0; mi < 2; ++mi) {
      v8bf lo, hi;
#pragma unroll
      for (int i = 0; i < 8; ++i) { lo[i] = (__bf16)0.f; hi[i] = (__bf16)0.f; }
      if (aval[mi]) {
        lo = *(const v8bf*)(arow[mi] + k0 + hf * 8);
        hi = *(const v8bf*)(arow[mi] + k0 + 16 + hf * 8);
      }
      af[mi] = cat8(lo, hi);
    }
#pragma unroll
    for (int ni = 0; ni < 2; ++ni) {
      b1f[ni] = load_b_frag_g(bcol1[ni], k0, hf);
      b3f[ni] = load_b_frag_g(bcol3[ni], k0, hf);
    }
#pragma unroll
    for (int mi = 0; mi < 2; ++mi)
#pragma unroll
      for (int ni = 0; ni < 2; ++ni) {
        accG[mi][ni] = __builtin_amdgcn_wmma_f32_16x16x32_bf16(
            false, af[mi], false, b1f[ni], (short)0, accG[mi][ni], false, false);
        accU[mi][ni] = __builtin_amdgcn_wmma_f32_16x16x32_bf16(
            false, af[mi], false, b3f[ni], (short)0, accU[mi][ni], false, false);
      }
  }

  // C/D layout: M = r + 8*(lane/16), N = lane%16
#pragma unroll
  for (int mi = 0; mi < 2; ++mi)
#pragma unroll
    for (int ni = 0; ni < 2; ++ni)
#pragma unroll
      for (int r = 0; r < 8; ++r) {
        int m = mt * TM + wm * 32 + mi * 16 + hf * 8 + r;
        int n = n0 + wn * 32 + ni * 16 + l16;
        float g = accG[mi][ni][r], u = accU[mi][ni][r];
        float hv = (g / (1.f + __expf(-g))) * u;
        hact[(size_t)(base + m) * Ii + n] = (__bf16)hv;
      }
}

// ---------------------------------------------------------------------------
// 5) FFN2: Y = hact @ W2 (fp32 out to pooled ybuf). LDS-free loop.
// ---------------------------------------------------------------------------
__global__ __launch_bounds__(256) void ffn2_kernel(
    const __bf16* __restrict__ hact, const __bf16* __restrict__ w2t,
    const int* __restrict__ row_base, const int* __restrict__ kept,
    float* __restrict__ ybuf)
{
  const int e  = blockIdx.z;
  const int nk = kept[e];
  const int mt = blockIdx.y;
  if (mt * TM >= nk) return;
  const int base = row_base[e];
  const int n0 = blockIdx.x * TN;

  const int tid = threadIdx.x, lane = tid & 31, wv = tid >> 5;
  const int wm = wv >> 2, wn = wv & 3;
  const int hf = lane >> 4, l16 = lane & 15;

  const __bf16* arow[2];
#pragma unroll
  for (int mi = 0; mi < 2; ++mi) {
    int m = base + mt * TM + wm * 32 + mi * 16 + l16;
    arow[mi] = hact + (size_t)m * Ii;
  }
  const __bf16* bcol[2];
#pragma unroll
  for (int ni = 0; ni < 2; ++ni) {
    int n = n0 + wn * 32 + ni * 16 + l16;
    bcol[ni] = w2t + ((size_t)e * Hh + n) * Ii;
  }

  v8f acc[2][2];
#pragma unroll
  for (int mi = 0; mi < 2; ++mi)
#pragma unroll
    for (int ni = 0; ni < 2; ++ni)
#pragma unroll
      for (int i = 0; i < 8; ++i) acc[mi][ni][i] = 0.f;

  for (int k0 = 0; k0 < Ii; k0 += TK) {
    v16bf af[2], bf[2];
#pragma unroll
    for (int mi = 0; mi < 2; ++mi) af[mi] = load_a_frag_g(arow[mi], k0, hf);
#pragma unroll
    for (int ni = 0; ni < 2; ++ni) bf[ni] = load_b_frag_g(bcol[ni], k0, hf);
#pragma unroll
    for (int mi = 0; mi < 2; ++mi)
#pragma unroll
      for (int ni = 0; ni < 2; ++ni)
        acc[mi][ni] = __builtin_amdgcn_wmma_f32_16x16x32_bf16(
            false, af[mi], false, bf[ni], (short)0, acc[mi][ni], false, false);
  }

#pragma unroll
  for (int mi = 0; mi < 2; ++mi)
#pragma unroll
    for (int ni = 0; ni < 2; ++ni)
#pragma unroll
      for (int r = 0; r < 8; ++r) {
        int m = mt * TM + wm * 32 + mi * 16 + hf * 8 + r;
        int n = n0 + wn * 32 + ni * 16 + l16;
        ybuf[(size_t)(base + m) * Hh + n] = acc[mi][ni][r];
      }
}

// ---------------------------------------------------------------------------
// 6) Combine: out[t,h] = sum_k w_clamped[t,k] * ybuf[slot_row[t,k], h]
// ---------------------------------------------------------------------------
__global__ __launch_bounds__(256) void combine_kernel(
    const float* __restrict__ ybuf, const int* __restrict__ slot_row,
    const float* __restrict__ wcl, float* __restrict__ out)
{
  const int t = blockIdx.x;
  const int r0 = slot_row[t * Kk + 0], r1 = slot_row[t * Kk + 1];
  const float w0 = wcl[t * Kk + 0], w1v = wcl[t * Kk + 1];
  for (int h = threadIdx.x; h < Hh; h += 256) {
    float a = 0.f;
    if (r0 >= 0) a += w0  * ybuf[(size_t)r0 * Hh + h];
    if (r1 >= 0) a += w1v * ybuf[(size_t)r1 * Hh + h];
    out[(size_t)t * Hh + h] = a;
  }
}

// ---------------------------------------------------------------------------
// 7) Aux loss
// ---------------------------------------------------------------------------
__global__ __launch_bounds__(256) void aux_kernel(
    const float* __restrict__ gate_probs, const int* __restrict__ kept,
    const float* __restrict__ droppedp, float* __restrict__ out_aux)
{
  __shared__ float simp[256 * Ee];
  __shared__ float stot[Ee];
  const int tid = threadIdx.x;
  float a[Ee];
#pragma unroll
  for (int e = 0; e < Ee; ++e) a[e] = 0.f;
  for (int t = tid; t < T; t += 256)
#pragma unroll
    for (int e = 0; e < Ee; ++e) a[e] += gate_probs[(size_t)t * Ee + e];
#pragma unroll
  for (int e = 0; e < Ee; ++e) simp[tid * Ee + e] = a[e];
  __syncthreads();
  if (tid < Ee) {
    float s = 0.f;
    for (int i = 0; i < 256; ++i) s += simp[i * Ee + tid];
    stot[tid] = s;
  }
  __syncthreads();
  if (tid == 0) {
    float aux = 0.f;
    for (int e = 0; e < Ee; ++e) {
      float usage = (float)kept[e] / (float)(T * Kk);
      float imp = stot[e] / (float)T;
      aux += usage * imp;
    }
    aux *= (float)Ee;
    float d = *droppedp;
    if (d > 0.f) aux += d / (float)T * 0.1f;
    out_aux[0] = fminf(aux, 1.0f) * 0.001f;
  }
}

// ---------------------------------------------------------------------------
extern "C" void kernel_launch(void* const* d_in, const int* in_sizes, int n_in,
                              void* d_out, int out_size, void* d_ws, size_t ws_size,
                              hipStream_t stream) {
  (void)in_sizes; (void)n_in; (void)out_size; (void)ws_size;
  const float* x  = (const float*)d_in[0];
  const float* wg = (const float*)d_in[1];
  const float* w1 = (const float*)d_in[2];
  const float* w3 = (const float*)d_in[3];
  const float* w2 = (const float*)d_in[4];
  float* out = (float*)d_out;

  char* ws = (char*)d_ws;
  size_t o = 0;
  auto alloc = [&](size_t bytes) -> char* {
    char* p = ws + o;
    o = (o + bytes + 255) & ~(size_t)255;
    return p;
  };
  float*  gate_probs = (float*) alloc((size_t)T * Ee * sizeof(float));
  int*    topk_idx   = (int*)   alloc((size_t)T * Kk * sizeof(int));
  float*  w_clamped  = (float*) alloc((size_t)T * Kk * sizeof(float));
  int*    slot_row   = (int*)   alloc((size_t)T * Kk * sizeof(int));
  int*    token_pool = (int*)   alloc((size_t)POOL * sizeof(int));
  int*    row_base   = (int*)   alloc(Ee * sizeof(int));
  int*    kept       = (int*)   alloc(Ee * sizeof(int));
  float*  dropped    = (float*) alloc(sizeof(float));
  __bf16* xbf        = (__bf16*)alloc((size_t)T * Hh * sizeof(__bf16));
  __bf16* w1t        = (__bf16*)alloc((size_t)Ee * Hh * Ii * sizeof(__bf16)); // [E][I][H]
  __bf16* w3t        = (__bf16*)alloc((size_t)Ee * Hh * Ii * sizeof(__bf16)); // [E][I][H]
  __bf16* w2t        = (__bf16*)alloc((size_t)Ee * Ii * Hh * sizeof(__bf16)); // [E][H][I]
  __bf16* hact       = (__bf16*)alloc((size_t)POOL * Ii * sizeof(__bf16));
  float*  ybuf       = (float*) alloc((size_t)POOL * Hh * sizeof(float));

  // Precision/layout pre-pass (bf16, K-major for WMMA fragments)
  cvt_x_kernel<<<(T * Hh) / (256 * 8), 256, 0, stream>>>(x, xbf);
  transpose_w_kernel<<<dim3(Ii / 64, Hh / 64, Ee), 256, 0, stream>>>(w1, w1t, Hh, Ii);
  transpose_w_kernel<<<dim3(Ii / 64, Hh / 64, Ee), 256, 0, stream>>>(w3, w3t, Hh, Ii);
  transpose_w_kernel<<<dim3(Hh / 64, Ii / 64, Ee), 256, 0, stream>>>(w2, w2t, Ii, Hh);

  routing_kernel<<<T / 8, 256, 0, stream>>>(x, wg, gate_probs, topk_idx, w_clamped);
  dispatch_kernel<<<1, 256, 0, stream>>>(topk_idx, slot_row, token_pool, row_base,
                                         kept, dropped);
  ffn1_kernel<<<dim3(Ii / TN, MAXMT, Ee), 256, 0, stream>>>(xbf, w1t, w3t, token_pool,
                                                            row_base, kept, hact);
  ffn2_kernel<<<dim3(Hh / TN, MAXMT, Ee), 256, 0, stream>>>(hact, w2t, row_base,
                                                            kept, ybuf);
  combine_kernel<<<T, 256, 0, stream>>>(ybuf, slot_row, w_clamped, out);
  aux_kernel<<<1, 256, 0, stream>>>(gate_probs, kept, dropped, out + (size_t)T * Hh);
}
